// BufferAttend1d_12859132084338
// MI455X (gfx1250) — matrix operand order repaired and checked by reading, not compiled, not run on hardware
//
#include <hip/hip_runtime.h>

#define B_   8
#define Q_   4096
#define K_   4096
#define DIN  256
#define D_   64
#define SCALE 0.125f
#define FILLV (-1024.0f)

typedef __attribute__((ext_vector_type(16))) __bf16 v16bf;
typedef __attribute__((ext_vector_type(8)))  __bf16 v8bf;
typedef __attribute__((ext_vector_type(4)))  __bf16 v4bf;
typedef __attribute__((ext_vector_type(8)))  float  v8f;
typedef __attribute__((ext_vector_type(8)))  unsigned v8u;

// Pack two f32 into two truncated bf16 with ONE v_perm_b32:
// result = { lo = a[31:16], hi = b[31:16] }.
// v_perm sel bytes: 0-3 pick src1 bytes, 4-7 pick src0 bytes (src0 = high dword).
static __device__ __forceinline__ unsigned pktrunc(float a, float b) {
  return __builtin_amdgcn_perm(__builtin_bit_cast(unsigned, b),
                               __builtin_bit_cast(unsigned, a), 0x07060302u);
}

// single f32 -> truncated bf16 (1 shift; folds into d16-hi stores)
static __device__ __forceinline__ __bf16 bftrunc(float f) {
  return __builtin_bit_cast(
      __bf16, (unsigned short)(__builtin_bit_cast(unsigned, f) >> 16));
}

static __device__ __forceinline__ v8f wmma_bf16(v16bf a, v16bf b, v8f c) {
  // D = A(16x32 bf16) * B(32x16 bf16) + C(16x16 f32)
  return __builtin_amdgcn_wmma_f32_16x16x32_bf16(false, a, false, b, (short)0, c,
                                                 false, false);
}

// A-matrix 16x32 bf16 layout (ISA 7.12.2): lane l holds row m=l&15;
// half h=l>>4: element e -> K = h*16 + e (contiguous, not the ISA table order).
// f32 source loader: 4x b128 loads + 8x v_perm for the whole 16-elem operand.
static __device__ __forceinline__ v16bf loadA_f32(const float* rowp, int h) {
  v8u t;
#pragma unroll
  for (int g = 0; g < 4; ++g) {
    const int ks = (h << 4) + (g << 2);
    float4 f = *(const float4*)(rowp + ks);
    t[g * 2 + 0] = pktrunc(f.x, f.y);
    t[g * 2 + 1] = pktrunc(f.z, f.w);
  }
  return __builtin_bit_cast(v16bf, t);
}

// Same A layout, source already bf16 (global or LDS): 4x b64 loads, no VALU.
static __device__ __forceinline__ v16bf loadA_bf16(const __bf16* rowp, int h) {
  v16bf r;
#pragma unroll
  for (int g = 0; g < 4; ++g) {
    const int ks = (h << 4) + (g << 2);
    v4bf f = *(const v4bf*)(rowp + ks);
#pragma unroll
    for (int i = 0; i < 4; ++i) r[g * 4 + i] = f[i];
  }
  return r;
}

// B operand k order used here: lane l holds column n=l&15; lanes 0-15 hold
// K=0..15, lanes 16-31 hold K=16..31, contiguous along K.
// p points at element K = h*16 of the column's backing (K-contiguous) row.
static __device__ __forceinline__ v16bf loadB_bf16_contig(const __bf16* p) {
  v8bf lo = *(const v8bf*)p;
  v8bf hi = *(const v8bf*)(p + 8);
  v16bf r;
#pragma unroll
  for (int i = 0; i < 8; ++i) { r[i] = lo[i]; r[i + 8] = hi[i]; }
  return r;
}

static __device__ __forceinline__ v16bf loadB_f32_contig(const float* p) {
  v8u t;
#pragma unroll
  for (int g = 0; g < 4; ++g) {
    float4 f = *(const float4*)(p + g * 4);
    t[g * 2 + 0] = pktrunc(f.x, f.y);
    t[g * 2 + 1] = pktrunc(f.z, f.w);
  }
  return __builtin_bit_cast(v16bf, t);
}

// ---------------------------------------------------------------------------
// Kernel 1: projections.
// z=0: Qp[row,n] = x@Wk^T+bk  (row-major bf16)
// z=1: Kp[row,n] = buffer@Wk^T+bk (row-major bf16)
// z=2: Vt[b][n][k] = (buffer@Wv^T+bv) TRANSPOSED per batch -> attention
//      stage-3 B-tiles are K-contiguous (pure b128 loads, no gather).
// ---------------------------------------------------------------------------
__global__ void proj_kernel(const float* __restrict__ x, const float* __restrict__ buf,
                            const float* __restrict__ Wk, const float* __restrict__ bk,
                            const float* __restrict__ Wv, const float* __restrict__ bv,
                            __bf16* __restrict__ Qp, __bf16* __restrict__ Kp,
                            __bf16* __restrict__ Vt) {
  const int z = blockIdx.z;
  const float* src  = (z == 0) ? x  : buf;
  const float* W    = (z == 2) ? Wv : Wk;
  const float* bias = (z == 2) ? bv : bk;

  const int lane = threadIdx.x & 31;
  const int wave = threadIdx.x >> 5;   // 0..3 = n-tile
  const int l16  = lane & 15;
  const int h    = lane >> 4;
  const int nbase = wave * 16;
  const long long row0 = (long long)blockIdx.x * 16;

  const float* arow = src + (row0 + l16) * DIN;            // A: row m = l16
  const float* brow = W + (long long)(nbase + l16) * DIN;  // B col n = weight row n

  v8f acc = {};
#pragma unroll
  for (int kk = 0; kk < DIN; kk += 32) {
    v16bf a = loadA_f32(arow + kk, h);
    v16bf b = loadB_f32_contig(brow + kk + h * 16);
    acc = wmma_bf16(a, b, acc);
  }
  const float bia = bias[nbase + l16];
  const int n = nbase + l16;
  if (z == 2) {
    // transposed store: Vt[b][n][k]
#pragma unroll
    for (int r = 0; r < 8; ++r) {
      const long long grow = row0 + r + 8 * h;   // global key row
      const long long bb = grow >> 12;           // /4096
      const long long kk = grow & (K_ - 1);
      Vt[(bb * D_ + n) * K_ + kk] = bftrunc(acc[r] + bia);
    }
  } else {
    __bf16* dst = (z == 0) ? Qp : Kp;
#pragma unroll
    for (int r = 0; r < 8; ++r) {
      const int m = r + 8 * h;                   // C/D: VGPR r, half h -> row M
      dst[(row0 + m) * D_ + n] = bftrunc(acc[r] + bia);
    }
  }
}

// ---------------------------------------------------------------------------
// Kernel 2: fused attention for a 16-query tile.
// 256 threads (8 wave32); 128 KB dynamic LDS = 16x4096 bf16 logit/prob tile
// (2 blocks co-resident per 320 KB WGP).
//  Stage 1: logits = QK^T*scale (+mask) via WMMA -> bf16 LDS
//  Stage 2: row softmax (f32 math), f32 probs streamed to HBM as float2,
//           bf16 probs repacked into LDS with v_perm
//  Stage 3: read = P@V via WMMA; A-tiles load straight from bf16 LDS (b64),
//           B-tiles from L2-resident transposed Vt (b128) -- zero cvt VALU
// ---------------------------------------------------------------------------
__global__ void attn_kernel(const __bf16* __restrict__ Qp, const __bf16* __restrict__ Kp,
                            const __bf16* __restrict__ Vt, const int* __restrict__ mask,
                            float* __restrict__ probs, float* __restrict__ readout) {
  extern __shared__ float smem[];
  __bf16* s_pb = (__bf16*)smem;              // 16 * 4096 bf16 = 128 KB

  const int b    = blockIdx.x >> 8;          // 4096/16 = 256 q-tiles per batch
  const int qt   = blockIdx.x & 255;
  const int row0 = qt * 16;

  const int lane = threadIdx.x & 31;
  const int wave = threadIdx.x >> 5;         // 0..7
  const int l16  = lane & 15;
  const int h    = lane >> 4;

  const __bf16* Qb  = Qp + ((long long)b * Q_ + row0) * D_;
  const __bf16* Kb  = Kp + (long long)b * K_ * D_;
  const __bf16* Vtb = Vt + (long long)b * D_ * K_;
  const int*    mb  = mask + b * K_;

  // ---- Stage 1: logits tile -> bf16 LDS ----
  const __bf16* qrow = Qb + l16 * D_;
  const v16bf a0 = loadA_bf16(qrow + 0,  h);   // head-dim 0..31
  const v16bf a1 = loadA_bf16(qrow + 32, h);   // head-dim 32..63

  for (int nt = wave; nt < K_ / 16; nt += 8) {
    const int n0 = nt * 16;
    const __bf16* krow = Kb + (long long)(n0 + l16) * D_;  // B col n = key row n
    __builtin_prefetch(krow + 8 * 16 * D_, 0, 1);          // next iteration's keys
    v16bf b0 = loadB_bf16_contig(krow + h * 16);
    v16bf b1 = loadB_bf16_contig(krow + 32 + h * 16);
    v8f acc = {};
    acc = wmma_bf16(a0, b0, acc);
    acc = wmma_bf16(a1, b1, acc);
    const int n  = n0 + l16;
    const int mv = mb[n];                       // nonzero -> masked out
#pragma unroll
    for (int r = 0; r < 8; ++r) {
      const int m = r + 8 * h;
      s_pb[m * K_ + n] = bftrunc(mv ? FILLV : acc[r] * SCALE);
    }
  }
  __syncthreads();

  // ---- Stage 2: softmax per row (2 rows per wave); packed b32 LDS traffic ----
#pragma unroll
  for (int i = 0; i < 2; ++i) {
    const int m = wave * 2 + i;
    unsigned* rowu = (unsigned*)s_pb + m * (K_ / 2);   // 2 bf16 per word
    float mx = -3.402823466e38f;
    for (int j = lane; j < K_ / 2; j += 32) {
      const unsigned u = rowu[j];
      const float f0 = __builtin_bit_cast(float, u << 16);
      const float f1 = __builtin_bit_cast(float, u & 0xffff0000u);
      mx = fmaxf(mx, fmaxf(f0, f1));
    }
#pragma unroll
    for (int off = 16; off > 0; off >>= 1) mx = fmaxf(mx, __shfl_xor(mx, off, 32));
    float sum = 0.0f;
    for (int j = lane; j < K_ / 2; j += 32) {
      const unsigned u = rowu[j];
      const float f0 = __builtin_bit_cast(float, u << 16);
      const float f1 = __builtin_bit_cast(float, u & 0xffff0000u);
      sum += __expf(f0 - mx) + __expf(f1 - mx);
    }
#pragma unroll
    for (int off = 16; off > 0; off >>= 1) sum += __shfl_xor(sum, off, 32);
    const float inv = 1.0f / sum;
    float* prow = probs + ((long long)b * Q_ + row0 + m) * K_;
    for (int j = lane; j < K_ / 2; j += 32) {
      const unsigned u = rowu[j];
      const float f0 = __builtin_bit_cast(float, u << 16);
      const float f1 = __builtin_bit_cast(float, u & 0xffff0000u);
      const float p0 = __expf(f0 - mx) * inv;
      const float p1 = __expf(f1 - mx) * inv;
      float2 pv; pv.x = p0; pv.y = p1;
      *(float2*)(prow + 2 * j) = pv;    // f32 probs: the single 512 MiB HBM write
      rowu[j] = pktrunc(p0, p1);        // bf16 probs for stage 3 (one v_perm)
    }
  }
  __syncthreads();

  // ---- Stage 3: read = P @ V ; each wave owns a 512-key slice ----
  v8f acc[4] = {};
  const int kbase0 = wave * (K_ / 8);
  for (int ks = 0; ks < K_ / 8; ks += 32) {
    const v16bf a = loadA_bf16(s_pb + l16 * K_ + kbase0 + ks, h);  // pure ds_load
#pragma unroll
    for (int nt = 0; nt < 4; ++nt) {
      const __bf16* vcol = Vtb + (long long)(nt * 16 + l16) * K_ + kbase0 + ks;
      v16bf bv = loadB_bf16_contig(vcol + h * 16);   // pure b128 global loads
      acc[nt] = wmma_bf16(a, bv, acc[nt]);
    }
  }
  __syncthreads();

  // cross-wave reduction of 16x64 partials (reuse LDS)
  float* s_part = smem;   // 8 waves * 1024 floats = 32 KB
#pragma unroll
  for (int nt = 0; nt < 4; ++nt) {
#pragma unroll
    for (int r = 0; r < 8; ++r) {
      const int m = r + 8 * h;
      s_part[wave * 1024 + m * 64 + nt * 16 + l16] = acc[nt][r];
    }
  }
  __syncthreads();
  for (int idx = threadIdx.x; idx < 1024; idx += 256) {
    float s = 0.0f;
#pragma unroll
    for (int w = 0; w < 8; ++w) s += s_part[w * 1024 + idx];
    const int m = idx >> 6, n = idx & 63;
    readout[((long long)b * Q_ + row0 + m) * D_ + n] = s;
  }
}

// ---------------------------------------------------------------------------
extern "C" void kernel_launch(void* const* d_in, const int* in_sizes, int n_in,
                              void* d_out, int out_size, void* d_ws, size_t ws_size,
                              hipStream_t stream) {
  (void)in_sizes; (void)n_in; (void)out_size; (void)ws_size;
  const float* x   = (const float*)d_in[0];
  const float* buf = (const float*)d_in[1];
  const int*   msk = (const int*)d_in[2];
  const float* Wk  = (const float*)d_in[3];
  const float* bk  = (const float*)d_in[4];
  const float* Wv  = (const float*)d_in[5];
  const float* bv  = (const float*)d_in[6];

  float* probs   = (float*)d_out;
  float* readout = probs + (long long)B_ * Q_ * K_;

  __bf16* Qp = (__bf16*)d_ws;                       // 4 MB
  __bf16* Kp = Qp + (long long)B_ * Q_ * D_;        // 4 MB
  __bf16* Vt = Kp + (long long)B_ * K_ * D_;        // 4 MB (transposed V)

  dim3 pgrid((B_ * Q_) / 16, 1, 3);
  proj_kernel<<<pgrid, dim3(128, 1, 1), 0, stream>>>(x, buf, Wk, bk, Wv, bv,
                                                     Qp, Kp, Vt);

  const size_t shbytes = (size_t)16 * K_ * sizeof(__bf16);  // 128 KB of 320 KB/WGP
  attn_kernel<<<dim3(B_ * (Q_ / 16), 1, 1), dim3(256, 1, 1), shbytes, stream>>>(
      Qp, Kp, Vt, msk, probs, readout);
}
